// MetaworldSACMixtureMHCriticNetwork_52536039964916
// MI455X (gfx1250) — compile-verified, matrix operand order checked by reading, not run on hardware
//
#include <hip/hip_runtime.h>

// ---------------------------------------------------------------------------
// MetaworldSACMixtureMHCriticNetwork fused forward for gfx1250 (MI455X).
// Math path: bf16 WMMA (v_wmma_f32_16x16x32_bf16) with f32 accumulation.
// Data path: double-buffered GLOBAL_LOAD_ASYNC_TO_LDS_B128 staging with
// in-order ASYNCcnt pipelining (s_wait_asynccnt 9 -> one full compute
// iteration of latency cover per slab).
// ---------------------------------------------------------------------------

typedef __attribute__((ext_vector_type(16))) __bf16 v16bf;
typedef __attribute__((ext_vector_type(8)))  float  v8f;
typedef unsigned short u16;

#define BATCH 8192
#define FDIM  1024
#define NEXP  8
#define NCTX  10
#define DIN   43
#define KPAD  64      // layer-0 K padded 43 -> 64
#define BM3   32      // batch rows per block in layer-1 kernel
#define KC    32      // k-step (one WMMA K)
#define LDP   40      // LDS row stride (elems): 16B-aligned rows, conflict-free
#define NSTEPS (NEXP * (FDIM / KC))   // 256 flattened (expert, k-step) steps

__device__ __forceinline__ u16 f2bf(float x) {
  unsigned int u = __float_as_uint(x);
  u += 0x7fffu + ((u >> 16) & 1u);       // round-to-nearest-even
  return (u16)(u >> 16);
}

union FragBf { v16bf f; uint4 q[2]; };

__device__ __forceinline__ v8f v8f_zero() {
  v8f z = {0.f, 0.f, 0.f, 0.f, 0.f, 0.f, 0.f, 0.f};
  return z;
}

// Async 16-byte copy global -> LDS (CDNA5 GLOBAL_LOAD_ASYNC_TO_LDS_B128,
// tracked by ASYNCcnt).  LDS operand is the wave-relative LDS byte address,
// which is the low 32 bits of the generic shared pointer (ISA 10.2).
__device__ __forceinline__ void async_g2l_b128(const u16* gsrc, u16* ldst) {
  asm volatile("global_load_async_to_lds_b128 %0, %1, off"
               :
               : "v"((unsigned int)(uintptr_t)ldst),
                 "v"((unsigned long long)(uintptr_t)gsrc)
               : "memory");
}

template <int N>
__device__ __forceinline__ void wait_async_le() {
  asm volatile("s_wait_asynccnt %0" ::"n"(N) : "memory");
}

// Load a 16x32 bf16 fragment from a row-major [16][ld] u16 tile.
// CDNA5 16-bit A/B layout: lane<16 -> row=lane, K {0..7,16..23};
// lane>=16 -> row=lane-16, K {8..15,24..31}.  `ld` multiple of 8, base 16B aligned.
__device__ __forceinline__ v16bf load_frag(const u16* base, int ld, int lane) {
  const int r  = lane & 15;
  const int kb = (lane & 16) >> 1;       // 0 or 8 elements
  const u16* p = base + r * ld + kb;
  FragBf u;
  u.q[0] = *(const uint4*)(p);
  u.q[1] = *(const uint4*)(p + 16);
  return u.f;
}

// ---------------------------------------------------------------------------
// Kernel 1: SA = bf16(concat(state, action)) padded to K=64.   [B][64]
// ---------------------------------------------------------------------------
__launch_bounds__(256)
__global__ void prep_sa_kernel(const float* __restrict__ st,
                               const float* __restrict__ ac,
                               u16* __restrict__ SAbf) {
  const int b = blockIdx.x * 256 + threadIdx.x;
  if (b >= BATCH) return;
  alignas(16) u16 tmp[KPAD];
  #pragma unroll
  for (int i = 0; i < 39; ++i) tmp[i] = f2bf(st[b * 39 + i]);
  #pragma unroll
  for (int i = 0; i < 4; ++i)  tmp[39 + i] = f2bf(ac[b * 4 + i]);
  #pragma unroll
  for (int i = DIN; i < KPAD; ++i) tmp[i] = 0;
  uint4* dst = (uint4*)(SAbf + (size_t)b * KPAD);
  const uint4* src = (const uint4*)tmp;
  #pragma unroll
  for (int q = 0; q < KPAD / 8; ++q) dst[q] = src[q];
}

// ---------------------------------------------------------------------------
// Kernel 2: W0t[e][n][k] = bf16(W0[e][k][n]), k>=43 zero-padded.  [E][F][64]
// ---------------------------------------------------------------------------
__launch_bounds__(256)
__global__ void prep_w0t_kernel(const float* __restrict__ W0,
                                u16* __restrict__ W0t) {
  const int idx = blockIdx.x * 256 + threadIdx.x;     // e*F*64 + n*64 + k
  if (idx >= NEXP * FDIM * KPAD) return;
  const int e = idx >> 16;
  const int n = (idx >> 6) & (FDIM - 1);
  const int k = idx & (KPAD - 1);
  W0t[idx] = (k < DIN) ? f2bf(W0[((size_t)e * DIN + k) * FDIM + n]) : (u16)0;
}

// ---------------------------------------------------------------------------
// Kernel 3: tiled transpose  W1t[e][n][k] = bf16(W1[e][k][n]).   [E][F][F]
// 64x64 tiles through LDS; coalesced on both global sides.
// ---------------------------------------------------------------------------
__launch_bounds__(256)
__global__ void prep_w1t_kernel(const float* __restrict__ W1,
                                u16* __restrict__ W1t) {
  __shared__ u16 Ts[64 * 72];                         // [k_local][n_local], pad 72
  const int t  = threadIdx.x;
  const int bx = blockIdx.x;                          // e*256 + kt*16 + nt
  const int e  = bx >> 8;
  const int kt = (bx >> 4) & 15;
  const int nt = bx & 15;

  const int tn  = t & 63;
  const int tk4 = t >> 6;                             // 0..3
  #pragma unroll
  for (int i = 0; i < 16; ++i) {
    const int kk = i * 4 + tk4;
    Ts[kk * 72 + tn] =
        f2bf(W1[((size_t)e * FDIM + kt * 64 + kk) * FDIM + nt * 64 + tn]);
  }
  __syncthreads();
  const int sk  = t & 63;
  const int sn4 = t >> 6;
  #pragma unroll
  for (int i = 0; i < 16; ++i) {
    const int nn = i * 4 + sn4;
    W1t[((size_t)e * FDIM + nt * 64 + nn) * FDIM + kt * 64 + sk] =
        Ts[sk * 72 + nn];
  }
}

// ---------------------------------------------------------------------------
// Kernel 4: layer 0.  H[e][b][f] = bf16(relu(SA @ W0[e] + b0[e])).
// Block = 256 threads = 8 waves; wave nj owns cols [nj*128, nj*128+128);
// block owns 16 batch rows of one expert.  K = 64 (2 WMMA k-steps).
// ---------------------------------------------------------------------------
__launch_bounds__(256)
__global__ void layer0_kernel(const u16* __restrict__ SAbf,
                              const u16* __restrict__ W0t,
                              const float* __restrict__ b0,
                              u16* __restrict__ H) {
  const int tid  = threadIdx.x;
  const int lane = tid & 31;
  const int nj   = tid >> 5;                          // 0..7
  const int bt   = blockIdx.x & (BATCH / 16 - 1);     // 512 row tiles
  const int e    = blockIdx.x >> 9;
  const int rowBase = bt * 16;

  v8f acc[8];
  #pragma unroll
  for (int nf = 0; nf < 8; ++nf) acc[nf] = v8f_zero();

  const u16* Arow = SAbf + (size_t)rowBase * KPAD;
  const u16* W0e  = W0t + (size_t)e * FDIM * KPAD;

  #pragma unroll
  for (int k0 = 0; k0 < KPAD; k0 += KC) {
    v16bf a = load_frag(Arow + k0, KPAD, lane);
    #pragma unroll
    for (int nf = 0; nf < 8; ++nf) {
      v16bf b = load_frag(W0e + (size_t)(nj * 128 + nf * 16) * KPAD + k0,
                          KPAD, lane);
      acc[nf] = __builtin_amdgcn_wmma_f32_16x16x32_bf16(
          false, a, false, b, (short)0, acc[nf], false, false);
    }
  }

  #pragma unroll
  for (int nf = 0; nf < 8; ++nf) {
    const int ncol = nj * 128 + nf * 16 + (lane & 15);
    const float bias = b0[e * FDIM + ncol];
    #pragma unroll
    for (int r = 0; r < 8; ++r) {
      const int row = rowBase + r + ((lane >> 4) << 3);
      const float h = fmaxf(acc[nf][r] + bias, 0.f);
      H[((size_t)(e * BATCH + row)) * FDIM + ncol] = f2bf(h);
    }
  }
}

// ---------------------------------------------------------------------------
// Stage one k-slab (step s) with async global->LDS copies.
// Every wave issues exactly 9 async instructions:
//   8 for the W1 slab [1024][KC] and 1 (partial-EXEC) for the H slab [32][KC].
// ---------------------------------------------------------------------------
__device__ __forceinline__ void stage_slab(const u16* __restrict__ W1t,
                                           const u16* __restrict__ H,
                                           int rowBase, int s,
                                           u16* BsBuf, u16* AsBuf, int tid) {
  const int e  = s >> 5;
  const int k0 = (s & 31) * KC;
  const u16* W1e  = W1t + (size_t)e * FDIM * FDIM + k0;
  const u16* Hrow = H + ((size_t)(e * BATCH + rowBase)) * FDIM + k0;
  #pragma unroll
  for (int j = 0; j < 8; ++j) {
    const int uidx = tid + j * 512;
    const int n = uidx >> 2, q = uidx & 3;
    async_g2l_b128(W1e + (size_t)n * FDIM + q * 8, BsBuf + n * LDP + q * 8);
  }
  if ((tid & 3) == 0) {                 // one partial-EXEC op in every wave
    const int idx = tid >> 2;           // 0..127
    const int rw = idx >> 2, q = idx & 3;
    async_g2l_b128(Hrow + (size_t)rw * FDIM + q * 8, AsBuf + rw * LDP + q * 8);
  }
}

// ---------------------------------------------------------------------------
// Kernel 5: layer 1 + mixture + context-routed head (the 137 GFLOP kernel).
// Block = 512 threads = 16 waves laid out 2(M) x 8(N); block owns 32 batch
// rows x all 1024 features, so mixing over experts and the head reduction
// stay in-block (deterministic, no atomics).  K-slabs are double-buffered
// via async-to-LDS: while step s computes, step s+1's DMA is in flight.
// ---------------------------------------------------------------------------
__launch_bounds__(512)
__global__ void layer1_mix_head_kernel(const u16* __restrict__ H,
                                       const u16* __restrict__ W1t,
                                       const float* __restrict__ b1,
                                       const float* __restrict__ Wte,
                                       const float* __restrict__ Wh,
                                       const float* __restrict__ bh,
                                       const int* __restrict__ cidx,
                                       float* __restrict__ out) {
  __shared__ u16   Bs[2][FDIM * LDP];                 // 2 x 80 KB W1 slabs
  __shared__ u16   As[2][BM3 * LDP];                  // 2 x 2.5 KB H slabs
  __shared__ float qpart[8][BM3];

  const int tid  = threadIdx.x;
  const int lane = tid & 31;
  const int wid  = tid >> 5;
  const int mi   = wid >> 3;                          // 0..1  (16-row slice)
  const int nj   = wid & 7;                           // 0..7  (128-col slice)
  const int rowBase = blockIdx.x * BM3;

  // context ids for this wave's 8 accumulator rows
  int cv[8];
  const int rlo = rowBase + mi * 16 + ((lane >> 4) << 3);
  #pragma unroll
  for (int r = 0; r < 8; ++r) cv[r] = cidx[rlo + r];

  v8f mixed[8];
  #pragma unroll
  for (int nf = 0; nf < 8; ++nf) mixed[nf] = v8f_zero();
  v8f acc[8];

  // prologue: kick off slab 0
  stage_slab(W1t, H, rowBase, 0, Bs[0], As[0], tid);

  for (int s = 0; s < NSTEPS; ++s) {
    const int e = s >> 5;
    if ((s & 31) == 0) {
      #pragma unroll
      for (int nf = 0; nf < 8; ++nf) acc[nf] = v8f_zero();
    }

    // issue next slab's DMA, then wait only for the current slab (in-order
    // ASYNCcnt: <=9 outstanding means slab s has fully landed).
    if (s + 1 < NSTEPS) {
      stage_slab(W1t, H, rowBase, s + 1, Bs[(s + 1) & 1], As[(s + 1) & 1], tid);
      wait_async_le<9>();
    } else {
      wait_async_le<0>();
    }
    __syncthreads();                                  // publish buffer s&1

    const u16* Bb = Bs[s & 1];
    const u16* Ab = As[s & 1];
    const v16bf a = load_frag(Ab + (mi * 16) * LDP, LDP, lane);
    #pragma unroll
    for (int nf = 0; nf < 8; ++nf) {
      const v16bf b = load_frag(Bb + (nj * 128 + nf * 16) * LDP, LDP, lane);
      acc[nf] = __builtin_amdgcn_wmma_f32_16x16x32_bf16(
          false, a, false, b, (short)0, acc[nf], false, false);
    }
    __syncthreads();   // all reads of buffer s&1 done -> safe to overwrite

    if ((s & 31) == 31) {
      // expert epilogue: bias + relu, then mixture-weighted accumulation
      float wv[8];
      #pragma unroll
      for (int r = 0; r < 8; ++r) wv[r] = Wte[e * NCTX + cv[r]];
      #pragma unroll
      for (int nf = 0; nf < 8; ++nf) {
        const int ncol = nj * 128 + nf * 16 + (lane & 15);
        const float bias = b1[e * FDIM + ncol];
        #pragma unroll
        for (int r = 0; r < 8; ++r) {
          const float h = fmaxf(acc[nf][r] + bias, 0.f);
          mixed[nf][r] += wv[r] * h;
        }
      }
    }
  }

  // relu(mixed) then per-row dot with context head Wh[c[b]]
  float part[8];
  #pragma unroll
  for (int r = 0; r < 8; ++r) part[r] = 0.f;
  #pragma unroll
  for (int nf = 0; nf < 8; ++nf) {
    const int ncol = nj * 128 + nf * 16 + (lane & 15);
    #pragma unroll
    for (int r = 0; r < 8; ++r) {
      const float m = fmaxf(mixed[nf][r], 0.f);
      part[r] += m * Wh[(size_t)cv[r] * FDIM + ncol];
    }
  }
  // reduce across the 16 lanes that share a row
  #pragma unroll
  for (int r = 0; r < 8; ++r) {
    #pragma unroll
    for (int off = 8; off >= 1; off >>= 1)
      part[r] += __shfl_xor(part[r], off, 16);
  }
  if ((lane & 15) == 0) {
    const int rb = mi * 16 + ((lane >> 4) << 3);
    #pragma unroll
    for (int r = 0; r < 8; ++r) qpart[nj][rb + r] = part[r];
  }
  __syncthreads();
  // deterministic in-order reduction across the 8 N-waves + bias head
  if (tid < BM3) {
    float s = 0.f;
    #pragma unroll
    for (int j = 0; j < 8; ++j) s += qpart[j][tid];
    const int cr = cidx[rowBase + tid];
    out[rowBase + tid] = s + bh[cr];
  }
}

// ---------------------------------------------------------------------------
extern "C" void kernel_launch(void* const* d_in, const int* in_sizes, int n_in,
                              void* d_out, int out_size, void* d_ws,
                              size_t ws_size, hipStream_t stream) {
  (void)in_sizes; (void)n_in; (void)out_size; (void)ws_size;
  const float* st  = (const float*)d_in[0];
  const float* ac  = (const float*)d_in[1];
  const int*   c   = (const int*)d_in[2];
  const float* W0  = (const float*)d_in[3];
  const float* b0  = (const float*)d_in[4];
  const float* W1  = (const float*)d_in[5];
  const float* b1  = (const float*)d_in[6];
  const float* Wte = (const float*)d_in[7];
  const float* Wh  = (const float*)d_in[8];
  const float* bh  = (const float*)d_in[9];
  float* out = (float*)d_out;

  // workspace layout (bf16), ~153 MB total
  u16* SAbf = (u16*)d_ws;                             //  1.0 MB  [B][64]
  u16* W0t  = SAbf + (size_t)BATCH * KPAD;            //  1.0 MB  [E][F][64]
  u16* W1t  = W0t  + (size_t)NEXP * FDIM * KPAD;      // 16.8 MB  [E][F][F]
  u16* H    = W1t  + (size_t)NEXP * FDIM * FDIM;      // 134 MB   [E][B][F]

  prep_sa_kernel<<<BATCH / 256, 256, 0, stream>>>(st, ac, SAbf);
  prep_w0t_kernel<<<(NEXP * FDIM * KPAD) / 256, 256, 0, stream>>>(W0, W0t);
  prep_w1t_kernel<<<NEXP * 16 * 16, 256, 0, stream>>>(W1, W1t);
  layer0_kernel<<<(BATCH / 16) * NEXP, 256, 0, stream>>>(SAbf, W0t, b0, H);
  layer1_mix_head_kernel<<<BATCH / BM3, 512, 0, stream>>>(H, W1t, b1, Wte, Wh,
                                                          bh, c, out);
}
